// GenericClassNet_13022340842269
// MI455X (gfx1250) — matrix operand order repaired and checked by reading, not compiled, not run on hardware
//
#include <hip/hip_runtime.h>
#include <hip/hip_bf16.h>

#define ROWPAD 136   // 16B-aligned, conflict-free LDS row stride (in bf16 elems)

typedef __bf16 bf16x16 __attribute__((ext_vector_type(16)));
typedef __bf16 bf16x8  __attribute__((ext_vector_type(8)));
typedef float  f32x8   __attribute__((ext_vector_type(8)));

// ---------------- structs passed by value ----------------
struct LayerD { int w_off; int b_off; int kt; int relu; };
struct ChainArgs {
  const float* in0; const float* in1; const float* resid; float* out;
  const float* ew; const __bf16* wbase; const float* bbase; const float* w15;
  int n; int nlayers; int mode; int pad;
  LayerD layers[16];
};
struct PrepEntry {
  const float* W; const float* b; const float* g; const float* be;
  const float* m; const float* v; int w_off; int b_off; int kt; int mode;
};
struct PrepArgs { int nmat; int pad; PrepEntry e[58]; };

// ---------------- param prep: fold BN, pack weights into WMMA B layout ----
__global__ __launch_bounds__(256) void gcn_prep(PrepArgs P, __bf16* wdst, float* bdst, float* ewdst) {
  PrepEntry E = P.e[blockIdx.x];
  __shared__ float s[64], t[64];
  int tid = threadIdx.x;
  if (tid < 64) {
    float sc = 1.f, tb = 0.f;
    if (E.g) { float iv = rsqrtf(E.v[tid] + 1e-5f); sc = E.g[tid] * iv; tb = E.be[tid] - E.m[tid] * sc; }
    s[tid] = sc; t[tid] = tb;
  }
  __syncthreads();
  if (E.mode == 0) {
    if (tid < 64) bdst[E.b_off + tid] = E.b[tid] * s[tid] + t[tid];
    int total = 2048 * E.kt;             // 4 ntiles * kt * 32 lanes * 16 elems
    for (int idx = tid; idx < total; idx += 256) {
      int i = idx & 15, lane = (idx >> 4) & 31, k = (idx >> 9) % E.kt, j = idx / (512 * E.kt);
      int N = j * 16 + (lane & 15);
      int K = k * 32 + (i & 7) + ((lane & 16) ? 8 : 0) + ((i & 8) ? 16 : 0);
      float w = E.W[K * 64 + N] * s[N];
      wdst[E.w_off + ((j * E.kt + k) * 32 + lane) * 16 + i] = (__bf16)w;
    }
  } else if (E.mode == 1) {              // bn_layer scale/bias vectors
    if (tid < 64) { ewdst[E.b_off + tid] = s[tid]; ewdst[E.b_off + 64 + tid] = t[tid]; }
  } else {                               // final predict head 64->1
    if (tid < 64) bdst[E.b_off + tid] = E.W[tid];
    if (tid == 0) bdst[E.b_off + 64] = E.b[0];
  }
}

// ---------------- small utility kernels ----------------
__global__ void gcn_zero(float* p, int n) {
  int i = blockIdx.x * blockDim.x + threadIdx.x;
  if (i < n) p[i] = 0.f;
}
__global__ void gcn_scale(const float* h, float* t, const float* eps, int n) {
  long i = (long)blockIdx.x * blockDim.x + threadIdx.x;
  if (i < (long)n * 16) {
    float e = 1.f + eps[0];
    float4 v = ((const float4*)h)[i];
    float4 o; o.x = v.x * e; o.y = v.y * e; o.z = v.z * e; o.w = v.w * e;
    ((float4*)t)[i] = o;
  }
}
__global__ void gcn_edge(const float* h, float* t, const int* src, const int* dst, int E) {
  long tid = (long)blockIdx.x * blockDim.x + threadIdx.x;
  int e = (int)(tid >> 4);
  if (e >= E) return;
  int c = (int)(tid & 15) * 4;
  long sN = src[e], dN = dst[e];
  float4 v = *(const float4*)(h + sN * 64 + c);
  float* p = t + dN * 64 + c;
  atomicAdd(p + 0, v.x); atomicAdd(p + 1, v.y); atomicAdd(p + 2, v.z); atomicAdd(p + 3, v.w);
}
__global__ void gcn_pool(const float* hp, const int* batch, float* sums, float* cnt, int n) {
  long tid = (long)blockIdx.x * blockDim.x + threadIdx.x;
  int r = (int)(tid >> 4);
  if (r >= n) return;
  int c = (int)(tid & 15) * 4;
  int b = batch[r];
  float4 v = *(const float4*)(hp + (long)r * 64 + c);
  float* p = sums + b * 64 + c;
  atomicAdd(p + 0, v.x); atomicAdd(p + 1, v.y); atomicAdd(p + 2, v.z); atomicAdd(p + 3, v.w);
  if (c == 0) atomicAdd(cnt + b, 1.f);
}
// divide sums by counts and pack p_rep [B,64] into WMMA B layout, K padded to 128
__global__ void gcn_packprep(const float* sums, const float* cnt, __bf16* dst, int Bb) {
  int idx = blockIdx.x * blockDim.x + threadIdx.x;
  if (idx >= 4 * 4 * 32 * 16) return;
  int i = idx & 15, lane = (idx >> 4) & 31, k = (idx >> 9) & 3, j = idx >> 11;
  int N = j * 16 + (lane & 15);
  int K = k * 32 + (i & 7) + ((lane & 16) ? 8 : 0) + ((i & 8) ? 16 : 0);
  float v = 0.f;
  if (K < Bb) v = sums[K * 64 + N] / fmaxf(cnt[K], 1.f);
  dst[((j * 4 + k) * 32 + lane) * 16 + i] = (__bf16)v;
}

// ---------------- WMMA building blocks ----------------
template<int KT>
__device__ inline void do_layer(const __bf16* my, const __bf16* wb, const float* bb,
                                int w_off, int b_off, int relu, int lane, f32x8 (&acc)[4]) {
  int half = lane >> 4, row = lane & 15;
  bf16x16 a[KT];
#pragma unroll
  for (int k = 0; k < KT; k++) {
    bf16x8 lo = *(const bf16x8*)&my[row * ROWPAD + k * 32 + half * 8];
    bf16x8 hi = *(const bf16x8*)&my[row * ROWPAD + k * 32 + half * 8 + 16];
    a[k] = __builtin_shufflevector(lo, hi, 0,1,2,3,4,5,6,7,8,9,10,11,12,13,14,15);
  }
#pragma unroll
  for (int j = 0; j < 4; j++) {
    f32x8 c = {0.f,0.f,0.f,0.f,0.f,0.f,0.f,0.f};
#pragma unroll
    for (int k = 0; k < KT; k++) {
      bf16x16 b = *(const bf16x16*)&wb[w_off + ((j * KT + k) * 32 + lane) * 16];
      c = __builtin_amdgcn_wmma_f32_16x16x32_bf16(false, a[k], false, b, (short)0, c, false, false);
    }
    float bias = bb[b_off + j * 16 + row];
#pragma unroll
    for (int v = 0; v < 8; v++) {
      float x = c[v] + bias;
      if (relu) x = fmaxf(x, 0.f);
      c[v] = x;
    }
    acc[j] = c;
  }
}
__device__ inline void store_lds(__bf16* my, const f32x8 (&acc)[4], int lane) {
  int half = lane >> 4, row = lane & 15;
#pragma unroll
  for (int j = 0; j < 4; j++)
#pragma unroll
    for (int v = 0; v < 8; v++)
      my[(v + half * 8) * ROWPAD + j * 16 + row] = (__bf16)acc[j][v];
}
__device__ inline void stage64(__bf16* my, const float* src, long rowbase, int n, int lane, int col_base) {
  int srow = lane >> 1, scol = (lane & 1) * 32;
  long r = rowbase + srow; if (r >= n) r = 0;
  const float4* p = (const float4*)(src + r * 64 + scol);
#pragma unroll
  for (int q = 0; q < 8; q++) {
    float4 f = p[q];
    int c = col_base + scol + q * 4;
    my[srow * ROWPAD + c + 0] = (__bf16)f.x;
    my[srow * ROWPAD + c + 1] = (__bf16)f.y;
    my[srow * ROWPAD + c + 2] = (__bf16)f.z;
    my[srow * ROWPAD + c + 3] = (__bf16)f.w;
  }
}

// ---------------- fused MLP chain (all GEMM stacks) ----------------
__global__ __launch_bounds__(256) void gcn_chain(ChainArgs A) {
  __shared__ __bf16 lds[8 * 16 * ROWPAD];
  int lane = threadIdx.x & 31, wave = threadIdx.x >> 5;
  __bf16* my = lds + wave * 16 * ROWPAD;
  long rowbase = (long)blockIdx.x * 128 + wave * 16;

  stage64(my, A.in0, rowbase, A.n, lane, 0);
  if (A.in1) stage64(my, A.in1, rowbase, A.n, lane, 64);

  f32x8 acc[4];
  for (int l = 0; l < A.nlayers; l++) {
    LayerD L = A.layers[l];
    if (L.kt == 4) do_layer<4>(my, A.wbase, A.bbase, L.w_off, L.b_off, L.relu, lane, acc);
    else           do_layer<2>(my, A.wbase, A.bbase, L.w_off, L.b_off, L.relu, lane, acc);
    bool last = (l == A.nlayers - 1);
    if (!last || A.mode == 1) store_lds(my, acc, lane);
  }

  int row = lane & 15, half = lane >> 4;
  if (A.mode == 1) {           // final 64->1 dot + sigmoid
    float s = 0.f;
#pragma unroll
    for (int c = 0; c < 32; c++)
      s += (float)my[row * ROWPAD + half * 32 + c] * A.w15[half * 32 + c];
    s += __shfl_xor(s, 16, 32);
    if (half == 0) {
      long r = rowbase + row;
      if (r < A.n) A.out[r] = 1.f / (1.f + __expf(-(s + A.w15[64])));
    }
  } else {                     // 64-wide output, optional bn_layer + residual
#pragma unroll
    for (int j = 0; j < 4; j++) {
      int col = j * 16 + row;
#pragma unroll
      for (int v = 0; v < 8; v++) {
        long r = rowbase + v + half * 8;
        if (r < A.n) {
          float x = acc[j][v];
          if (A.ew)    x = fmaxf(x * A.ew[col] + A.ew[64 + col], 0.f);
          if (A.resid) x += A.resid[r * 64 + col];
          A.out[r * 64 + col] = x;
        }
      }
    }
  }
}

// ---------------- ext[N,B] @ p_rep[B,64] with K padded to 128 ----------------
__global__ __launch_bounds__(256) void gcn_extgemm(const float* ext, const __bf16* prp,
                                                   float* G, int n, int Bb) {
  __shared__ __bf16 lds[8 * 16 * ROWPAD];
  int lane = threadIdx.x & 31, wave = threadIdx.x >> 5;
  __bf16* my = lds + wave * 16 * ROWPAD;
  long rowbase = (long)blockIdx.x * 128 + wave * 16;

  int srow = lane >> 1, cb = (lane & 1) * 64;
  long r0 = rowbase + srow; if (r0 >= n) r0 = 0;
  const float* src = ext + r0 * Bb;
#pragma unroll
  for (int q = 0; q < 16; q++) {
    int c = cb + q * 4;
    float4 f; f.x = f.y = f.z = f.w = 0.f;
    if (c < Bb) f = *(const float4*)(src + c);
    my[srow * ROWPAD + c + 0] = (__bf16)f.x;
    my[srow * ROWPAD + c + 1] = (__bf16)f.y;
    my[srow * ROWPAD + c + 2] = (__bf16)f.z;
    my[srow * ROWPAD + c + 3] = (__bf16)f.w;
  }

  int half = lane >> 4, row = lane & 15;
  bf16x16 a[4];
#pragma unroll
  for (int k = 0; k < 4; k++) {
    bf16x8 lo = *(const bf16x8*)&my[row * ROWPAD + k * 32 + half * 8];
    bf16x8 hi = *(const bf16x8*)&my[row * ROWPAD + k * 32 + half * 8 + 16];
    a[k] = __builtin_shufflevector(lo, hi, 0,1,2,3,4,5,6,7,8,9,10,11,12,13,14,15);
  }
#pragma unroll
  for (int j = 0; j < 4; j++) {
    f32x8 c = {0.f,0.f,0.f,0.f,0.f,0.f,0.f,0.f};
#pragma unroll
    for (int k = 0; k < 4; k++) {
      bf16x16 b = *(const bf16x16*)&prp[((j * 4 + k) * 32 + lane) * 16];
      c = __builtin_amdgcn_wmma_f32_16x16x32_bf16(false, a[k], false, b, (short)0, c, false, false);
    }
#pragma unroll
    for (int v = 0; v < 8; v++) {
      long r = rowbase + v + half * 8;
      if (r < n) G[r * 64 + j * 16 + row] = c[v];
    }
  }
}

// ---------------- host orchestration ----------------
extern "C" void kernel_launch(void* const* d_in, const int* in_sizes, int n_in,
                              void* d_out, int out_size, void* d_ws, size_t ws_size,
                              hipStream_t stream) {
  const float* X     = (const float*)d_in[0];
  const float* Xp    = (const float*)d_in[1];
  const int* src_g   = (const int*)d_in[2];
  const int* dst_g   = (const int*)d_in[3];
  const int* src_p   = (const int*)d_in[4];
  const int* dst_p   = (const int*)d_in[5];
  const int* batch_p = (const int*)d_in[6];
  const float* ext   = (const float*)d_in[7];
  const int Ng = in_sizes[0] / 64, Np = in_sizes[1] / 64;
  const int Eg = in_sizes[2], Ep = in_sizes[4];
  const int Bb = in_sizes[7] / Ng;

  const int IDX_EMB_G = 8, IDX_EMB_P = 10;
  const int IDX_GIN_G = 12, IDX_GIN_P = 80;
  const int IDX_MLP_G = 148, IDX_MLP_P = 192, IDX_PRED = 236;

  char* ws = (char*)d_ws;
  __bf16* Wpk = (__bf16*)ws;
  float* Bias = (float*)(ws + (1u << 20));
  float* EW   = (float*)(ws + (1u << 20) + (64u << 10));
  float* SUMS = (float*)(ws + (1u << 20) + (128u << 10));
  float* CNT  = SUMS + 6400;
  __bf16* PRP = (__bf16*)(ws + (1u << 20) + (192u << 10));
  size_t big = 2ull << 20;
  float* G   = (float*)(ws + big);
  float* HG0 = (float*)(ws + big + 28ull * (1ull << 20));
  float* HG1 = (float*)(ws + big + 56ull * (1ull << 20));
  float* HP0 = (float*)(ws + big + 84ull * (1ull << 20));
  float* HP1 = (float*)(ws + big + 90ull * (1ull << 20));

  // ---- build prep table: 49 GEMM weights, 1 final head, 8 bn_layer vecs ----
  PrepArgs P{};
  int nm = 0, woff = 0, boff = 0;
  int WO[49], BO[49];
  auto addW = [&](int wi, int bi, int g0, int kt, int id) {
    PrepEntry& E = P.e[nm++];
    E.W = (const float*)d_in[wi]; E.b = (const float*)d_in[bi];
    if (g0 >= 0) {
      E.g = (const float*)d_in[g0];     E.be = (const float*)d_in[g0 + 1];
      E.m = (const float*)d_in[g0 + 2]; E.v  = (const float*)d_in[g0 + 3];
    } else { E.g = E.be = E.m = E.v = nullptr; }
    E.w_off = woff; E.b_off = boff; E.kt = kt; E.mode = 0;
    WO[id] = woff; BO[id] = boff;
    woff += 2048 * kt; boff += 64;
  };
  addW(IDX_EMB_G, IDX_EMB_G + 1, -1, 2, 0);
  for (int i = 0; i < 4; i++) {
    int b = IDX_GIN_G + 17 * i;
    addW(b + 1, b + 2, b + 5, 2, 1 + 2 * i);   // W0 fold bns[0]
    addW(b + 3, b + 4, b + 9, 2, 2 + 2 * i);   // W1 fold bn_apply
  }
  for (int l = 0; l < 8; l++)
    addW(IDX_MLP_G + 2 * l, IDX_MLP_G + 2 * l + 1, (l < 7) ? (IDX_MLP_G + 16 + 4 * l) : -1, 2, 9 + l);
  addW(IDX_EMB_P, IDX_EMB_P + 1, -1, 2, 17);
  for (int i = 0; i < 4; i++) {
    int b = IDX_GIN_P + 17 * i;
    addW(b + 1, b + 2, b + 5, 2, 18 + 2 * i);
    addW(b + 3, b + 4, b + 9, 2, 19 + 2 * i);
  }
  for (int l = 0; l < 8; l++)
    addW(IDX_MLP_P + 2 * l, IDX_MLP_P + 2 * l + 1, (l < 7) ? (IDX_MLP_P + 16 + 4 * l) : -1, 2, 26 + l);
  for (int l = 0; l < 15; l++)
    addW(IDX_PRED + 2 * l, IDX_PRED + 2 * l + 1, IDX_PRED + 32 + 4 * l, (l == 0) ? 4 : 2, 34 + l);
  int b15_off = boff;
  { // final predict layer 64->1 (mode 2)
    PrepEntry& E = P.e[nm++];
    E.W = (const float*)d_in[IDX_PRED + 30]; E.b = (const float*)d_in[IDX_PRED + 31];
    E.g = E.be = E.m = E.v = nullptr;
    E.w_off = 0; E.b_off = b15_off; E.kt = 0; E.mode = 2;
    boff += 65;
  }
  for (int i = 0; i < 8; i++) { // bn_layer vectors (mode 1)
    int b = ((i < 4) ? IDX_GIN_G + 17 * i : IDX_GIN_P + 17 * (i - 4)) + 13;
    PrepEntry& E = P.e[nm++];
    E.W = nullptr; E.b = nullptr;
    E.g = (const float*)d_in[b];     E.be = (const float*)d_in[b + 1];
    E.m = (const float*)d_in[b + 2]; E.v  = (const float*)d_in[b + 3];
    E.w_off = 0; E.b_off = 128 * i; E.kt = 0; E.mode = 1;
  }
  P.nmat = nm;

  gcn_prep<<<nm, 256, 0, stream>>>(P, Wpk, Bias, EW);
  gcn_zero<<<(6528 + 255) / 256, 256, 0, stream>>>(SUMS, 6400 + 128);

  auto chain = [&](const float* in0, const float* in1, const float* resid, float* out,
                   const float* ew, const float* w15, int n, int nl, int mode,
                   const int* ids, const int* rl) {
    ChainArgs C{};
    C.in0 = in0; C.in1 = in1; C.resid = resid; C.out = out; C.ew = ew;
    C.wbase = Wpk; C.bbase = Bias; C.w15 = w15; C.n = n; C.nlayers = nl; C.mode = mode;
    for (int l = 0; l < nl; l++) {
      C.layers[l].w_off = WO[ids[l]]; C.layers[l].b_off = BO[ids[l]];
      C.layers[l].kt = (ids[l] == 34) ? 4 : 2; C.layers[l].relu = rl[l];
    }
    gcn_chain<<<(n + 127) / 128, 256, 0, stream>>>(C);
  };

  // ---- graph branch ----
  { int id = 0, rl = 0; chain(X, nullptr, nullptr, HG0, nullptr, nullptr, Ng, 1, 0, &id, &rl); }
  float* hc = HG0; float* ht = HG1;
  for (int i = 0; i < 4; i++) {
    const float* eps = (const float*)d_in[IDX_GIN_G + 17 * i];
    gcn_scale<<<(unsigned)(((long)Ng * 16 + 255) / 256), 256, 0, stream>>>(hc, ht, eps, Ng);
    gcn_edge<<<(unsigned)(((long)Eg * 16 + 255) / 256), 256, 0, stream>>>(hc, ht, src_g, dst_g, Eg);
    int ids[2] = {1 + 2 * i, 2 + 2 * i}; int rl[2] = {1, 1};
    chain(ht, nullptr, hc, ht, EW + 128 * i, nullptr, Ng, 2, 0, ids, rl);
    float* tmp = hc; hc = ht; ht = tmp;
  }
  { int ids[8], rl[8]; for (int l = 0; l < 8; l++) { ids[l] = 9 + l; rl[l] = (l < 7); }
    chain(hc, nullptr, nullptr, ht, nullptr, nullptr, Ng, 8, 0, ids, rl); }
  float* Hg = ht;

  // ---- pattern branch ----
  { int id = 17, rl = 0; chain(Xp, nullptr, nullptr, HP0, nullptr, nullptr, Np, 1, 0, &id, &rl); }
  float* pc = HP0; float* pt = HP1;
  for (int i = 0; i < 4; i++) {
    const float* eps = (const float*)d_in[IDX_GIN_P + 17 * i];
    gcn_scale<<<(unsigned)(((long)Np * 16 + 255) / 256), 256, 0, stream>>>(pc, pt, eps, Np);
    gcn_edge<<<(unsigned)(((long)Ep * 16 + 255) / 256), 256, 0, stream>>>(pc, pt, src_p, dst_p, Ep);
    int ids[2] = {18 + 2 * i, 19 + 2 * i}; int rl[2] = {1, 1};
    chain(pt, nullptr, pc, pt, EW + 128 * (4 + i), nullptr, Np, 2, 0, ids, rl);
    float* tmp = pc; pc = pt; pt = tmp;
  }
  { int ids[8], rl[8]; for (int l = 0; l < 8; l++) { ids[l] = 26 + l; rl[l] = (l < 7); }
    chain(pc, nullptr, nullptr, pt, nullptr, nullptr, Np, 8, 0, ids, rl); }
  float* Hp = pt;

  // ---- mean pool -> p_rep -> ext @ p_rep -> predict ----
  gcn_pool<<<(unsigned)(((long)Np * 16 + 255) / 256), 256, 0, stream>>>(Hp, batch_p, SUMS, CNT, Np);
  gcn_packprep<<<32, 256, 0, stream>>>(SUMS, CNT, PRP, Bb);
  gcn_extgemm<<<(Ng + 127) / 128, 256, 0, stream>>>(ext, PRP, G, Ng, Bb);
  { int ids[15], rl[15]; for (int l = 0; l < 15; l++) { ids[l] = 34 + l; rl[l] = 1; }
    chain(Hg, G, nullptr, (float*)d_out, nullptr, Bias + b15_off, Ng, 15, 1, ids, rl); }
}
